// LinformerSelfAttention_87591563034719
// MI455X (gfx1250) — compile-verified, matrix operand order checked by reading, not run on hardware
//
#include <hip/hip_runtime.h>
#include <hip/hip_bf16.h>

// ---------------------------------------------------------------------------
// Linformer self-attention for MI455X (gfx1250), bf16 WMMA pipeline with
// async-to-LDS (ASYNCcnt) double-buffered GEMM staging.
// ---------------------------------------------------------------------------

typedef __bf16 bf16_t;
typedef __attribute__((ext_vector_type(16))) __bf16 v16bf;
typedef __attribute__((ext_vector_type(8)))  __bf16 v8bf;
typedef __attribute__((ext_vector_type(8)))  float  v8f;

union Frag { v16bf v; v8bf h[2]; };

#define DIMC      1024
#define SEQC      4096
#define NFEAT     256
#define NHEADS    16
#define HEADDIM   64
#define BATCHC    4

// Async copy of 16 bytes global -> LDS, tracked by ASYNCcnt.
__device__ __forceinline__ void async_copy_b128(const bf16_t* gsrc, bf16_t* ldst) {
  unsigned lds = (unsigned)(size_t)ldst;             // low 32 bits = LDS offset
  unsigned long long ga = (unsigned long long)gsrc;
  asm volatile("global_load_async_to_lds_b128 %0, %1, off"
               :: "v"(lds), "v"(ga) : "memory");
}
__device__ __forceinline__ void wait_async0() {
  asm volatile("s_wait_asynccnt 0" ::: "memory");
}

// ------------------------------ fp32 -> bf16 -------------------------------
__global__ void cvt_f32_to_bf16(const float* __restrict__ in,
                                bf16_t* __restrict__ out, int n4) {
  int i = blockIdx.x * blockDim.x + threadIdx.x;
  int stride = gridDim.x * blockDim.x;
  for (; i < n4; i += stride) {
    float4 f = ((const float4*)in)[i];
    out[4*i+0] = (bf16_t)f.x;
    out[4*i+1] = (bf16_t)f.y;
    out[4*i+2] = (bf16_t)f.z;
    out[4*i+3] = (bf16_t)f.w;
  }
}

// ------------------------------ WMMA GEMM ----------------------------------
// C[M,N] = alpha * (A @ B) (+ bias), A: (M,K) row-major or (K,M) if TRANS_A,
// B: (K,N) row-major. 128x128 block tile, K-step 32, 256 threads = 8 waves,
// each wave computes a 32x64 sub-tile as 2x4 WMMA 16x16 tiles.
// Double-buffered LDS; A tile staged via GLOBAL_LOAD_ASYNC_TO_LDS_B128.
#define BM 128
#define BN 128
#define BKK 32

template<bool TRANS_A, bool OUT_F32, bool ADD_BIAS>
__global__ __launch_bounds__(256)
void gemm_bf16_wmma(const bf16_t* __restrict__ A,
                    const bf16_t* __restrict__ B,
                    void* __restrict__ Cv,
                    const float* __restrict__ bias,
                    float alpha,
                    int M, int N, int K,
                    int lda, int ldb, int ldc,
                    long long sA, long long sB, long long sC) {
  __shared__ __attribute__((aligned(16))) bf16_t As[2][BM * BKK]; // As[r*32+k]
  __shared__ __attribute__((aligned(16))) bf16_t Bs[2][BN * BKK]; // Bs[n*32+k]

  const int tid  = threadIdx.x;
  const int wave = tid >> 5;
  const int lane = tid & 31;
  const int wm   = wave & 3;                  // 4 waves along M (32 rows each)
  const int wn   = wave >> 2;                 // 2 waves along N (64 cols each)
  const int lrow = lane & 15;
  const int klo  = (lane & 16) ? 8 : 0;       // K-chunk select per ISA layout
  const int roff = (lane & 16) ? 8 : 0;       // C-layout row offset

  const int m0 = blockIdx.y * BM;
  const int n0 = blockIdx.x * BN;

  A += (size_t)blockIdx.z * sA;
  B += (size_t)blockIdx.z * sB;

  v8f acc[2][4] = {};

  auto stage = [&](int k0, int buf) {
    // -------- A tile --------
    if (!TRANS_A) {
      // pure 2D block copy: async global -> LDS, no VGPR round trip
      for (int c = tid; c < (BM * BKK) / 8; c += 256) {
        int r  = c >> 2;
        int kk = (c & 3) << 3;
        async_copy_b128(A + (size_t)(m0 + r) * lda + k0 + kk,
                        &As[buf][r * BKK + kk]);
      }
    } else {
      for (int c = tid; c < (BM * BKK) / 8; c += 256) {
        int kk = c & 31;
        int r0 = (c >> 5) << 3;
        v8bf t = *(const v8bf*)(A + (size_t)(k0 + kk) * lda + m0 + r0);
        #pragma unroll
        for (int i = 0; i < 8; i++) As[buf][(r0 + i) * BKK + kk] = t[i];
      }
    }
    // -------- B tile (transpose into column-packed layout) --------
    for (int c = tid; c < (BN * BKK) / 8; c += 256) {
      int kk  = c & 31;
      int nn0 = (c >> 5) << 3;
      v8bf t = *(const v8bf*)(B + (size_t)(k0 + kk) * ldb + n0 + nn0);
      #pragma unroll
      for (int i = 0; i < 8; i++) Bs[buf][(nn0 + i) * BKK + kk] = t[i];
    }
  };

  const int nk = K / BKK;
  stage(0, 0);
  wait_async0();
  __syncthreads();

  for (int i = 0; i < nk; i++) {
    const int buf = i & 1;
    if (i + 1 < nk) stage((i + 1) * BKK, buf ^ 1);   // overlap with compute

    Frag afr[2], bfr[4];
    #pragma unroll
    for (int rt = 0; rt < 2; rt++) {
      int r = wm * 32 + rt * 16 + lrow;
      afr[rt].h[0] = *(const v8bf*)(&As[buf][r * BKK + klo]);
      afr[rt].h[1] = *(const v8bf*)(&As[buf][r * BKK + klo + 16]);
    }
    #pragma unroll
    for (int nt = 0; nt < 4; nt++) {
      int n = wn * 64 + nt * 16 + lrow;
      bfr[nt].h[0] = *(const v8bf*)(&Bs[buf][n * BKK + klo]);
      bfr[nt].h[1] = *(const v8bf*)(&Bs[buf][n * BKK + klo + 16]);
    }
    #pragma unroll
    for (int rt = 0; rt < 2; rt++)
      #pragma unroll
      for (int nt = 0; nt < 4; nt++)
        acc[rt][nt] = __builtin_amdgcn_wmma_f32_16x16x32_bf16(
            false, afr[rt].v, false, bfr[nt].v, (short)0, acc[rt][nt],
            false, false);

    wait_async0();        // next tile's async A copies have landed
    __syncthreads();      // + everyone done reading/writing this round
  }

  // -------- epilogue --------
  #pragma unroll
  for (int rt = 0; rt < 2; rt++) {
    #pragma unroll
    for (int nt = 0; nt < 4; nt++) {
      int col = n0 + wn * 64 + nt * 16 + lrow;
      float bv = ADD_BIAS ? bias[col] : 0.0f;
      #pragma unroll
      for (int j = 0; j < 8; j++) {
        int row = m0 + wm * 32 + rt * 16 + roff + j;
        float v = acc[rt][nt][j] * alpha + bv;
        size_t idx = (size_t)blockIdx.z * sC + (size_t)row * ldc + col;
        if (OUT_F32) ((float*)Cv)[idx] = v;
        else         ((bf16_t*)Cv)[idx] = (bf16_t)v;
      }
    }
  }
}

// --------------------------- fused attention -------------------------------
// Per block: one (batch, head, 64 q-rows) tile. 4 waves, 16 rows per wave.
// scores (16x256 f32) -> softmax -> P(bf16) -> out = P @ V_head (16x64).
__global__ __launch_bounds__(128)
void linformer_attn_wmma(const bf16_t* __restrict__ q,    // (B, SEQ, DIM), pre-scaled
                         const bf16_t* __restrict__ kp,   // (B, NFEAT, DIM)
                         const bf16_t* __restrict__ vp,   // (B, NFEAT, DIM)
                         bf16_t* __restrict__ out) {      // (B, SEQ, DIM)
  __shared__ __attribute__((aligned(16))) bf16_t Vt[HEADDIM * NFEAT]; // Vt[dh*256+kf]
  __shared__ __attribute__((aligned(16))) bf16_t P[4 * 16 * NFEAT];   // per-wave 16x256

  const int tid  = threadIdx.x;
  const int wave = tid >> 5;
  const int lane = tid & 31;
  const int lrow = lane & 15;
  const int klo  = (lane & 16) ? 8 : 0;
  const int roff = (lane & 16) ? 8 : 0;

  const int b  = blockIdx.z;
  const int h  = blockIdx.y;
  const int n0 = blockIdx.x * 64;
  const size_t hoff = (size_t)h * HEADDIM;

  // ---- stage V_head transposed into LDS ----
  for (int c = tid; c < (NFEAT * HEADDIM) / 8; c += 128) {
    int kf  = c >> 3;
    int dh0 = (c & 7) << 3;
    v8bf t = *(const v8bf*)(vp + (size_t)(b * NFEAT + kf) * DIMC + hoff + dh0);
    #pragma unroll
    for (int i = 0; i < 8; i++) Vt[(dh0 + i) * NFEAT + kf] = t[i];
  }
  __syncthreads();

  // ---- scores: q_tile (16x64) @ K_head^T (64x256) ----
  v8f s[16] = {};
  const int gr = n0 + wave * 16 + lrow;
  #pragma unroll
  for (int ds = 0; ds < HEADDIM; ds += 32) {
    Frag aq;
    const bf16_t* qrow = q + ((size_t)b * SEQC + gr) * DIMC + hoff + ds;
    aq.h[0] = *(const v8bf*)(qrow + klo);
    aq.h[1] = *(const v8bf*)(qrow + klo + 16);
    #pragma unroll
    for (int kt = 0; kt < 16; kt++) {
      Frag bk;  // B operand col = feature kf; contiguous along d in kp rows
      const bf16_t* krow =
          kp + (size_t)(b * NFEAT + kt * 16 + lrow) * DIMC + hoff + ds;
      bk.h[0] = *(const v8bf*)(krow + klo);
      bk.h[1] = *(const v8bf*)(krow + klo + 16);
      s[kt] = __builtin_amdgcn_wmma_f32_16x16x32_bf16(
          false, aq.v, false, bk.v, (short)0, s[kt], false, false);
    }
  }

  // ---- softmax over 256 features (rows live across 16-lane halves) ----
  bf16_t* Pw = &P[wave * 16 * NFEAT];
  #pragma unroll
  for (int j = 0; j < 8; j++) {
    float mx = -3.4e38f;
    #pragma unroll
    for (int kt = 0; kt < 16; kt++) mx = fmaxf(mx, s[kt][j]);
    #pragma unroll
    for (int off = 8; off >= 1; off >>= 1)
      mx = fmaxf(mx, __shfl_xor(mx, off, 32));
    float sum = 0.f;
    #pragma unroll
    for (int kt = 0; kt < 16; kt++) {
      float e = __expf(s[kt][j] - mx);
      s[kt][j] = e;
      sum += e;
    }
    #pragma unroll
    for (int off = 8; off >= 1; off >>= 1) sum += __shfl_xor(sum, off, 32);
    float inv = 1.0f / sum;
    #pragma unroll
    for (int kt = 0; kt < 16; kt++)
      Pw[(roff + j) * NFEAT + kt * 16 + lrow] = (bf16_t)(s[kt][j] * inv);
  }
  // Per-wave LDS buffer: LDS ops are in-order within a wave (DScnt), no
  // block barrier required before re-reading Pw.

  // ---- out_tile (16x64) = P (16x256) @ V_head (256x64) ----
  #pragma unroll
  for (int nt = 0; nt < 4; nt++) {
    v8f o = {};
    #pragma unroll
    for (int ks = 0; ks < NFEAT; ks += 32) {
      Frag ap, bv;
      ap.h[0] = *(const v8bf*)(&Pw[lrow * NFEAT + ks + klo]);
      ap.h[1] = *(const v8bf*)(&Pw[lrow * NFEAT + ks + klo + 16]);
      int dh = nt * 16 + lrow;
      bv.h[0] = *(const v8bf*)(&Vt[dh * NFEAT + ks + klo]);
      bv.h[1] = *(const v8bf*)(&Vt[dh * NFEAT + ks + klo + 16]);
      o = __builtin_amdgcn_wmma_f32_16x16x32_bf16(
          false, ap.v, false, bv.v, (short)0, o, false, false);
    }
    #pragma unroll
    for (int j = 0; j < 8; j++) {
      int row = n0 + wave * 16 + roff + j;
      out[((size_t)b * SEQC + row) * DIMC + hoff + nt * 16 + lrow] =
          (bf16_t)o[j];
    }
  }
}

// ------------------------------- launch ------------------------------------
extern "C" void kernel_launch(void* const* d_in, const int* in_sizes, int n_in,
                              void* d_out, int out_size, void* d_ws,
                              size_t ws_size, hipStream_t stream) {
  (void)in_sizes; (void)n_in; (void)out_size; (void)ws_size;

  const float* x   = (const float*)d_in[0];
  const float* Wq  = (const float*)d_in[1];
  const float* Wkv = (const float*)d_in[2];
  const float* Wp  = (const float*)d_in[3];
  const float* bp  = (const float*)d_in[4];
  const float* pk  = (const float*)d_in[5];
  const float* pv  = (const float*)d_in[6];
  float* out = (float*)d_out;

  const size_t nX   = (size_t)BATCHC * SEQC * DIMC;     // 16,777,216
  const size_t nWq  = (size_t)DIMC * DIMC;              // 1,048,576
  const size_t nWkv = (size_t)DIMC * 2 * DIMC;          // 2,097,152
  const size_t nPrj = (size_t)SEQC * NFEAT;             // 1,048,576

  char* w = (char*)d_ws;
  bf16_t* xb   = (bf16_t*)w;  w += nX   * 2;
  bf16_t* wqb  = (bf16_t*)w;  w += nWq  * 2;
  bf16_t* wkvb = (bf16_t*)w;  w += nWkv * 2;
  bf16_t* wpb  = (bf16_t*)w;  w += nWq  * 2;
  bf16_t* pkb  = (bf16_t*)w;  w += nPrj * 2;
  bf16_t* pvb  = (bf16_t*)w;  w += nPrj * 2;
  bf16_t* qb   = (bf16_t*)w;  w += nX   * 2;            // (B*SEQ, DIM)
  bf16_t* kvb  = (bf16_t*)w;  w += nX * 2 * 2;          // (B*SEQ, 2*DIM)
  bf16_t* kprj = (bf16_t*)w;  w += (size_t)BATCHC * NFEAT * DIMC * 2;
  bf16_t* vprj = (bf16_t*)w;  w += (size_t)BATCHC * NFEAT * DIMC * 2;
  bf16_t* attb = (bf16_t*)w;  w += nX * 2;              // attention output

  // ---- fp32 -> bf16 conversions ----
  dim3 cb(256);
  cvt_f32_to_bf16<<<dim3(4096), cb, 0, stream>>>(x,   xb,   (int)(nX   / 4));
  cvt_f32_to_bf16<<<dim3(512),  cb, 0, stream>>>(Wq,  wqb,  (int)(nWq  / 4));
  cvt_f32_to_bf16<<<dim3(512),  cb, 0, stream>>>(Wkv, wkvb, (int)(nWkv / 4));
  cvt_f32_to_bf16<<<dim3(512),  cb, 0, stream>>>(Wp,  wpb,  (int)(nWq  / 4));
  cvt_f32_to_bf16<<<dim3(512),  cb, 0, stream>>>(pk,  pkb,  (int)(nPrj / 4));
  cvt_f32_to_bf16<<<dim3(512),  cb, 0, stream>>>(pv,  pvb,  (int)(nPrj / 4));

  const int MB = BATCHC * SEQC;  // 16384 flattened rows
  dim3 gb(256);

  // ---- q = (x @ Wq) * dh^-0.5 ----
  gemm_bf16_wmma<false, false, false>
      <<<dim3(DIMC / BN, MB / BM, 1), gb, 0, stream>>>(
          xb, wqb, qb, nullptr, 0.125f, MB, DIMC, DIMC, DIMC, DIMC, DIMC,
          0, 0, 0);

  // ---- kv = x @ Wkv ----
  gemm_bf16_wmma<false, false, false>
      <<<dim3(2 * DIMC / BN, MB / BM, 1), gb, 0, stream>>>(
          xb, wkvb, kvb, nullptr, 1.0f, MB, 2 * DIMC, DIMC, DIMC, 2 * DIMC,
          2 * DIMC, 0, 0, 0);

  // ---- keys_proj[b]  = proj_k^T @ keys[b]   (256 x 1024, inner 4096) ----
  gemm_bf16_wmma<true, false, false>
      <<<dim3(DIMC / BN, NFEAT / BM, BATCHC), gb, 0, stream>>>(
          pkb, kvb, kprj, nullptr, 1.0f, NFEAT, DIMC, SEQC, NFEAT, 2 * DIMC,
          DIMC, 0, (long long)SEQC * 2 * DIMC, (long long)NFEAT * DIMC);

  // ---- values_proj[b] = proj_v^T @ values[b] (values = kv[:, DIM:]) ----
  gemm_bf16_wmma<true, false, false>
      <<<dim3(DIMC / BN, NFEAT / BM, BATCHC), gb, 0, stream>>>(
          pvb, kvb + DIMC, vprj, nullptr, 1.0f, NFEAT, DIMC, SEQC, NFEAT,
          2 * DIMC, DIMC, 0, (long long)SEQC * 2 * DIMC,
          (long long)NFEAT * DIMC);

  // ---- fused softmax(q K^T) V per head ----
  linformer_attn_wmma<<<dim3(SEQC / 64, NHEADS, BATCHC), dim3(128), 0,
                        stream>>>(qb, kprj, vprj, attb);

  // ---- final projection: out = attn_out @ Wproj + bproj (fp32 out) ----
  gemm_bf16_wmma<false, true, true>
      <<<dim3(DIMC / BN, MB / BM, 1), gb, 0, stream>>>(
          attb, wpb, out, bp, 1.0f, MB, DIMC, DIMC, DIMC, DIMC, DIMC,
          0, 0, 0);
}
